// RingAttentionV2Simple_73375221285555
// MI455X (gfx1250) — matrix operand order repaired and checked by reading, not compiled, not run on hardware
//
#include <hip/hip_runtime.h>
#include <math.h>

typedef __attribute__((ext_vector_type(16))) _Float16 vh16;
typedef __attribute__((ext_vector_type(2)))  _Float16 vh2;
typedef __attribute__((ext_vector_type(2)))  __fp16   vp2;
typedef __attribute__((ext_vector_type(8)))  float    vf8;

union AFrag { vh16 v; vh2 h[8]; };

// pack two f32 -> two f16 (v_cvt_pkrtz_f16_f32), bit-cast to _Float16 vector
__device__ __forceinline__ vh2 pkh(float a, float b) {
  vp2 t = __builtin_amdgcn_cvt_pkrtz(a, b);
  return __builtin_bit_cast(vh2, t);
}

#define B_   2
#define N_   2048
#define H_   16
#define D_   64
#define QT_  64     // queries per workgroup (4 waves x 16)
#define KT_  64     // keys per tile (doubles WMMA per softmax/barrier round)
#define KSTR 68     // kLds row stride (f16 elems): [key][d+pad]
#define VSTR 66     // vLds row stride (f16 elems): [d][key+pad]
#define PSTR 66     // pLds row stride (f16 elems): [m][key+pad]

template <int PAT>
__device__ __forceinline__ float swz(float x) {
  return __int_as_float(__builtin_amdgcn_ds_swizzle(__float_as_int(x), PAT));
}
// reductions across the 16-lane half-wave (xor masks < 16 stay within half)
__device__ __forceinline__ float redmax16(float x) {
  x = fmaxf(x, swz<0x041F>(x));   // SWAPX1
  x = fmaxf(x, swz<0x081F>(x));   // SWAPX2
  x = fmaxf(x, swz<0x101F>(x));   // SWAPX4
  x = fmaxf(x, swz<0x201F>(x));   // SWAPX8
  return x;
}
__device__ __forceinline__ float redsum16(float x) {
  x += swz<0x041F>(x);
  x += swz<0x081F>(x);
  x += swz<0x101F>(x);
  x += swz<0x201F>(x);
  return x;
}

__global__ __launch_bounds__(128)
void fa_fwd_causal(const float* __restrict__ Q, const float* __restrict__ K,
                   const float* __restrict__ V, float* __restrict__ O) {
  __shared__ __attribute__((aligned(16))) _Float16 kLds[KT_ * KSTR];     // 8704 B
  __shared__ __attribute__((aligned(16))) _Float16 vLds[D_ * VSTR];      // 8448 B
  __shared__ __attribute__((aligned(16))) _Float16 pLds[4][16 * PSTR];   // 8448 B

  const int qblk = blockIdx.x;
  const int h    = blockIdx.y;
  const int b    = blockIdx.z;
  const int tid  = threadIdx.x;
  const int wave = tid >> 5;
  const int lane = tid & 31;
  const int laneLo = lane & 15;
  const int hi   = lane >> 4;            // which 16-lane half

  const size_t ROW = (size_t)H_ * D_;    // stride between sequence positions
  const int qBase = qblk * QT_;
  const int qRow0 = qBase + wave * 16;   // first query row of this wave

  const float* qp = Q + (size_t)b * N_ * ROW + (size_t)(qRow0 + laneLo) * ROW + (size_t)h * D_;
  const float* kBase = K + (size_t)b * N_ * ROW + (size_t)h * D_;
  const float* vBase = V + (size_t)b * N_ * ROW + (size_t)h * D_;

  // ---- Load Q as two A-fragments (d 0..31, 32..63), scale folded in -------
  // A layout (16-bit, 16x32): lane = M (both halves), VGPR r holds K-pair at
  // d = (r&3)*2 + 8*hi + 16*(r>=4)  (+32*chunk)
  AFrag qa[2];
#pragma unroll
  for (int c = 0; c < 2; ++c) {
#pragma unroll
    for (int r = 0; r < 8; ++r) {
      const int d0 = ((r & 3) << 1) + (hi << 3) + ((r >> 2) << 4) + (c << 5);
      const float2 f = *(const float2*)(qp + d0);
      qa[c].h[r] = pkh(f.x * 0.125f, f.y * 0.125f);
    }
  }

  // ---- Flash state (C-layout rows: m = r + 8*hi) --------------------------
  vf8 o[4] = {{}, {}, {}, {}};
  float mrow[8], lrow[8];
#pragma unroll
  for (int r = 0; r < 8; ++r) { mrow[r] = -__builtin_inff(); lrow[r] = 0.0f; }

  const int nTiles = qblk + 1;           // causal: keys up to qBase+63
  const int sKey = tid >> 1;             // staging: key row 0..63 (2 thr/row)
  const int sD   = (tid & 1) << 5;       // staging: 32 d-elems per thread

  for (int t = 0; t < nTiles; ++t) {
    const int tk = t * KT_;
    const float* kr = kBase + (size_t)(tk + sKey) * ROW + sD;
    const float* vr = vBase + (size_t)(tk + sKey) * ROW + sD;

    __syncthreads();                     // prior tile's LDS reads done
    // ---- Stage K (row-major f16) and V (transposed [d][key]) -------------
#pragma unroll
    for (int j = 0; j < 8; ++j) {
      const float4 fk = *(const float4*)(kr + 4 * j);
      *(vh2*)&kLds[sKey * KSTR + sD + 4 * j    ] = pkh(fk.x, fk.y);
      *(vh2*)&kLds[sKey * KSTR + sD + 4 * j + 2] = pkh(fk.z, fk.w);
      const float4 fv = *(const float4*)(vr + 4 * j);
      vLds[(sD + 4 * j + 0) * VSTR + sKey] = (_Float16)fv.x;
      vLds[(sD + 4 * j + 1) * VSTR + sKey] = (_Float16)fv.y;
      vLds[(sD + 4 * j + 2) * VSTR + sKey] = (_Float16)fv.z;
      vLds[(sD + 4 * j + 3) * VSTR + sKey] = (_Float16)fv.w;
    }
    if (t + 1 < nTiles) {                // pull next tile toward L2/L0
      __builtin_prefetch(kr + (size_t)KT_ * ROW, 0, 1);
      __builtin_prefetch(vr + (size_t)KT_ * ROW, 0, 1);
    }
    __syncthreads();                     // staging visible to all waves

    if (tk <= qRow0 + 15) {              // wave-uniform: tile intersects causal range
      // ---- S = (Q*scale) @ K^T : four 16x16 column tiles, 2 k-steps each --
      // B layout (32x16): VGPR r = contraction pair at (r&3)*2 + 8*hi + 16*(r>=4); lanes = N,
      // half 0 holds contraction 0-7,16-23, half 1 holds 8-15,24-31.
      vf8 s[4];
#pragma unroll
      for (int g = 0; g < 4; ++g) {
        vf8 c = {};
#pragma unroll
        for (int st = 0; st < 2; ++st) {
          AFrag kb;
#pragma unroll
          for (int r = 0; r < 8; ++r)
            kb.h[r] = *(const vh2*)&kLds[(g * 16 + laneLo) * KSTR +
                                         (((r & 3) << 1) + (hi << 3) + ((r >> 2) << 4) + (st << 5))];
          c = __builtin_amdgcn_wmma_f32_16x16x32_f16(false, qa[st].v, false, kb.v,
                                                     (short)0, c, false, false);
        }
        s[g] = c;
      }

      // ---- Causal mask (C layout: row m = r+8*hi, col n = laneLo+16g) -----
      if (tk + KT_ - 1 > qRow0) {
#pragma unroll
        for (int g = 0; g < 4; ++g) {
          const int kcol = tk + (g << 4) + laneLo;
#pragma unroll
          for (int r = 0; r < 8; ++r) {
            const int qg = qRow0 + r + (hi << 3);
            s[g][r] = (kcol > qg) ? -__builtin_inff() : s[g][r];
          }
        }
      }

      // ---- Online softmax update ------------------------------------------
#pragma unroll
      for (int r = 0; r < 8; ++r) {
        float mx = fmaxf(fmaxf(s[0][r], s[1][r]), fmaxf(s[2][r], s[3][r]));
        mx = redmax16(mx);
        const float mn    = fmaxf(mrow[r], mx);
        const float alpha = __expf(mrow[r] - mn);
        const float p0 = __expf(s[0][r] - mn);
        const float p1 = __expf(s[1][r] - mn);
        const float p2 = __expf(s[2][r] - mn);
        const float p3 = __expf(s[3][r] - mn);
        const float rs = redsum16((p0 + p1) + (p2 + p3));
        lrow[r] = lrow[r] * alpha + rs;
        mrow[r] = mn;
        o[0][r] *= alpha; o[1][r] *= alpha; o[2][r] *= alpha; o[3][r] *= alpha;
        const int m = r + (hi << 3);
        pLds[wave][m * PSTR + laneLo     ] = (_Float16)p0;   // C->LDS
        pLds[wave][m * PSTR + laneLo + 16] = (_Float16)p1;
        pLds[wave][m * PSTR + laneLo + 32] = (_Float16)p2;
        pLds[wave][m * PSTR + laneLo + 48] = (_Float16)p3;
      }

      // ---- P back as two A-fragments (keys 0-31, 32-63) via per-wave LDS --
      AFrag pa[2];
#pragma unroll
      for (int c = 0; c < 2; ++c)
#pragma unroll
        for (int r = 0; r < 8; ++r)
          pa[c].h[r] = *(const vh2*)&pLds[wave][laneLo * PSTR +
                                                (((r & 3) << 1) + (hi << 3) +
                                                 ((r >> 2) << 4) + (c << 5))];

      // ---- O += P @ V : B-fragment = packed key-pairs from transposed V ---
#pragma unroll
      for (int f = 0; f < 4; ++f) {
#pragma unroll
        for (int c = 0; c < 2; ++c) {
          AFrag vb;
#pragma unroll
          for (int r = 0; r < 8; ++r)
            vb.h[r] = *(const vh2*)&vLds[((f << 4) + laneLo) * VSTR +
                                         (((r & 3) << 1) + (hi << 3) + ((r >> 2) << 4) + (c << 5))];
          o[f] = __builtin_amdgcn_wmma_f32_16x16x32_f16(false, pa[c].v, false, vb.v,
                                                        (short)0, o[f], false, false);
        }
      }
    }
  }

  // ---- Epilogue: normalize and store (fp32, [b, n, h, d]) -----------------
#pragma unroll
  for (int r = 0; r < 8; ++r) {
    const float inv = 1.0f / lrow[r];
    const int qg = qRow0 + r + (hi << 3);
    float* op = O + (size_t)b * N_ * ROW + (size_t)qg * ROW + (size_t)h * D_ + laneLo;
    op[0]  = o[0][r] * inv;
    op[16] = o[1][r] * inv;
    op[32] = o[2][r] * inv;
    op[48] = o[3][r] * inv;
  }
}

extern "C" void kernel_launch(void* const* d_in, const int* in_sizes, int n_in,
                              void* d_out, int out_size, void* d_ws, size_t ws_size,
                              hipStream_t stream) {
  const float* q = (const float*)d_in[0];
  const float* k = (const float*)d_in[1];
  const float* v = (const float*)d_in[2];
  float* out = (float*)d_out;
  (void)in_sizes; (void)n_in; (void)out_size; (void)d_ws; (void)ws_size;

  dim3 grid(N_ / QT_, H_, B_);   // (32, 16, 2)
  dim3 block(128);               // 4 waves of 32
  fa_fwd_causal<<<grid, block, 0, stream>>>(q, k, v, out);
}
